// WaveRNN_481036337284
// MI455X (gfx1250) — matrix operand order repaired and checked by reading, not compiled, not run on hardware
//
#include <hip/hip_runtime.h>
#include <hip/hip_bf16.h>
#include <math.h>

// ---------------- problem constants ----------------
#define B_    16
#define T_    4096
#define RNN_  896
#define HALF_ 448
#define G3_   2688          // 3*RNN
#define FC_   896
#define NC_   256
#define AUX_  64
#define DINP_ 160           // 147 padded to 32-multiple
#define M_    (B_*T_)       // 65536 flattened (b,t) rows
#define KC_   512           // HALF + AUX

#define NWG_  28            // GRU workgroups: 896/32 units
#define UU_   32            // hidden units per GRU workgroup
#define LDK_  904           // LDS row stride in halfs (896 + 8 pad -> breaks bank conflicts)
#define GRU_SMEM ((size_t)(96*LDK_*2 + 96*16*4))   // W_hh slice + gate staging = ~176KB

#if __has_builtin(__builtin_amdgcn_global_load_async_to_lds_b128)
#define HAVE_ASYNC_LDS 1
#endif

typedef __attribute__((ext_vector_type(16))) _Float16 v16h;
typedef __attribute__((ext_vector_type(8)))  _Float16 v8h;
typedef __attribute__((ext_vector_type(8)))  float    v8f;
typedef __attribute__((ext_vector_type(4)))  int      v4i;
typedef __attribute__((address_space(1))) v4i* gv4i_p;   // global b128 payload ptr
typedef __attribute__((address_space(3))) v4i* lv4i_p;   // LDS b128 payload ptr

__device__ inline v8f wmma_f16(v16h a, v16h b, v8f c) {
  // D = A(16x32 f16) x B(32x16 f16) + C(16x16 f32)
  return __builtin_amdgcn_wmma_f32_16x16x32_f16(false, a, false, b, (short)0, c, false, false);
}

// A-matrix fragment (16x32 f16), row-major source, lda in halfs.
// Lane L (0..15): M=L, K=k0..k0+15 contiguous; lane L+16: K=k0+16..k0+31.
__device__ inline v16h load_a16(const _Float16* A, int lda, int k0, int lane) {
  int m = lane & 15, hi = lane >> 4;
  const _Float16* p = A + (size_t)m * lda + k0 + hi * 16;
  return *(const v16h*)p;
}

// B-matrix fragment (32x16 f16) from W stored N-major over K (W[n][k]), ldk in halfs.
// Lane n (0..15): col N=n, K=k0..k0+15 contiguous; lane n+16: K=k0+16..k0+31.
__device__ inline v16h load_b16(const _Float16* W, int ldk, int n0, int k0, int lane) {
  const _Float16* p = W + (size_t)(n0 + (lane & 15)) * ldk + k0 + (lane >> 4) * 16;
  return *(const v16h*)p;   // 32B -> 2x b128
}

// ---------------- generic WMMA GEMM:  out = act(A[M,K] * W[N,K]^T + bias) ----------------
// One wave computes a 16x64 tile (4 accumulators); A frag reused across the 4 B frags.
__global__ __launch_bounds__(256) void gemm16(const _Float16* __restrict__ A,
                                              const _Float16* __restrict__ W,
                                              const float* __restrict__ bias,
                                              _Float16* __restrict__ out16,
                                              float* __restrict__ out32,
                                              int Mtiles, int Ngroups, int N, int K, int relu) {
  int lane = threadIdx.x & 31;
  int gw = blockIdx.x * (blockDim.x >> 5) + (threadIdx.x >> 5);
  if (gw >= Mtiles * Ngroups) return;              // uniform per wave: EXEC stays all-ones
  int mt = gw % Mtiles;
  int ng = gw / Mtiles;
  const _Float16* Arow = A + (size_t)mt * 16 * K;
  int n0 = ng * 64;

  v8f z = {0.f,0.f,0.f,0.f,0.f,0.f,0.f,0.f};
  v8f acc[4] = {z, z, z, z};
  for (int k0 = 0; k0 < K; k0 += 32) {
    v16h a = load_a16(Arow, K, k0, lane);
#pragma unroll
    for (int s = 0; s < 4; ++s) {
      v16h b = load_b16(W, K, n0 + s * 16, k0, lane);
      acc[s] = wmma_f16(a, b, acc[s]);
    }
  }
  int col16 = lane & 15, hi = lane >> 4;
#pragma unroll
  for (int s = 0; s < 4; ++s) {
    int col = n0 + s * 16 + col16;
    float bv = bias ? bias[col] : 0.f;
#pragma unroll
    for (int r = 0; r < 8; ++r) {
      int row = mt * 16 + r + hi * 8;              // C layout: VGPR r -> M=r (lo half), M=r+8 (hi half)
      float v = acc[s][r] + bv;
      if (relu) v = fmaxf(v, 0.f);
      if (out16) out16[(size_t)row * N + col] = (_Float16)v;
      if (out32) out32[(size_t)row * N + col] = v;
    }
  }
}

// ---------------- persistent GRU kernel ----------------
// 28 WGs x 192 threads (6 waves). WG w owns hidden units [32w, 32w+32): W_hh rows
// {j, 896+j, 1792+j} resident in LDS (96 rows x 896 f16, padded stride). Per step:
// wave i computes hp tile i (16 gate rows x 16 batch) via 28 WMMAs, stages into LDS,
// then all threads do the pointwise gates and write h_{t+1}; device-scope atomic barrier.
__global__ __launch_bounds__(192) void gru_kernel(const _Float16* __restrict__ Whh16,
                                                  const float* __restrict__ bhh,
                                                  const _Float16* __restrict__ xproj,
                                                  _Float16* __restrict__ hbuf,   // [2][16][896] f16
                                                  _Float16* __restrict__ Ac,     // [M][512]
                                                  _Float16* __restrict__ Af,     // [M][512]
                                                  float* __restrict__ hn_out,    // [16][896]
                                                  unsigned* __restrict__ ctr) {
  extern __shared__ char smem[];
  _Float16* wsl = (_Float16*)smem;                          // [96][LDK_]
  float* hpS = (float*)(smem + (size_t)96 * LDK_ * 2);      // [96][16]
  const int tid = threadIdx.x, wg = blockIdx.x;
  const int lane = tid & 31, wave = tid >> 5;

  // Preload this WG's W_hh slice into LDS (16B chunks).
  // Prefer CDNA5 register-bypassing async copy (ASYNCcnt) when the builtin exists.
  for (int c = tid; c < 96 * 112; c += blockDim.x) {
    int lr = c / 112, off = (c % 112) * 8;
    int gate = lr >> 5, u = lr & 31;
    int grow = gate * RNN_ + wg * UU_ + u;
    const _Float16* src = Whh16 + (size_t)grow * RNN_ + off;
    _Float16* dst = wsl + (size_t)lr * LDK_ + off;
#ifdef HAVE_ASYNC_LDS
    __builtin_amdgcn_global_load_async_to_lds_b128((gv4i_p)(void*)src, (lv4i_p)(void*)dst, 0, 0);
#else
    *(v8h*)dst = *(const v8h*)src;
#endif
  }
#ifdef HAVE_ASYNC_LDS
#if __has_builtin(__builtin_amdgcn_s_wait_asynccnt)
  __builtin_amdgcn_s_wait_asynccnt(0);
#endif
#endif
  __syncthreads();

  for (int t = 0; t < T_; ++t) {
    const _Float16* hcur = hbuf + (size_t)(t & 1) * B_ * RNN_;
    _Float16* hnext = hbuf + (size_t)((t + 1) & 1) * B_ * RNN_;

    // Prefetch this step's x_proj gate rows so they arrive during the WMMA phase
    // (global_prefetch_b8; x_proj reads otherwise stall the serial gate phase).
    if (tid < 48) {
      int b = tid & 15, gate = tid >> 4;
      const _Float16* pf = xproj + ((size_t)b * T_ + t) * G3_ + gate * RNN_ + wg * UU_;
      __builtin_prefetch(pf, 0, 1);
    }

    // hp = h_t @ Whh_slice^T  (one 16x16 tile per wave, K=896)
    v8f acc = {0.f,0.f,0.f,0.f,0.f,0.f,0.f,0.f};
    for (int k = 0; k < 28; ++k) {
      v16h a = load_a16(hcur, RNN_, k * 32, lane);
      v16h b = load_b16(wsl, LDK_, wave * 16, k * 32, lane);
      acc = wmma_f16(a, b, acc);
    }
    int col = lane & 15, hi = lane >> 4;
#pragma unroll
    for (int r = 0; r < 8; ++r)
      hpS[(wave * 16 + col) * 16 + r + hi * 8] = acc[r];    // hpS[local_gate_row][batch]
    __syncthreads();

    // Pointwise gates: 512 (u,b) pairs on 192 threads; u in lane dim for coalesced xproj reads.
    for (int idx = tid; idx < UU_ * B_; idx += blockDim.x) {
      int u = idx & 31, b = idx >> 5;
      int j = wg * UU_ + u;
      float hr = hpS[u * 16 + b]        + bhh[j];
      float hz = hpS[(32 + u) * 16 + b] + bhh[RNN_ + j];
      float hn = hpS[(64 + u) * 16 + b] + bhh[2 * RNN_ + j];
      size_t m = (size_t)b * T_ + t;
      const _Float16* xp = xproj + m * G3_;
      float xr = (float)xp[j], xz = (float)xp[RNN_ + j], xn = (float)xp[2 * RNN_ + j];
      float hprev = (float)hcur[(size_t)b * RNN_ + j];
      float rg = 1.f / (1.f + expf(-(xr + hr)));
      float zg = 1.f / (1.f + expf(-(xz + hz)));
      float ng = tanhf(xn + rg * hn);
      float hnew = (1.f - zg) * ng + zg * hprev;
      hnext[(size_t)b * RNN_ + j] = (_Float16)hnew;
      if (j < HALF_) Ac[m * KC_ + j] = (_Float16)hnew;
      else           Af[m * KC_ + (j - HALF_)] = (_Float16)hnew;
      if (t == T_ - 1) hn_out[(size_t)b * RNN_ + j] = hnew;
    }
    __syncthreads();

    // Device-scope grid barrier (28 co-resident WGs), s_sleep backoff while spinning.
    __threadfence();
    if (tid == 0) {
      __hip_atomic_fetch_add(ctr, 1u, __ATOMIC_ACQ_REL, __HIP_MEMORY_SCOPE_AGENT);
      unsigned target = (unsigned)NWG_ * (unsigned)(t + 1);
      while (__hip_atomic_load(ctr, __ATOMIC_ACQUIRE, __HIP_MEMORY_SCOPE_AGENT) < target)
        __builtin_amdgcn_s_sleep(2);
    }
    __syncthreads();
  }
}

// ---------------- small prep / epilogue kernels ----------------
__global__ void cvt_pad(const float* __restrict__ src, _Float16* __restrict__ dst,
                        int rows, int sc, int dc) {
  size_t i = (size_t)blockIdx.x * blockDim.x + threadIdx.x;
  if (i >= (size_t)rows * dc) return;
  int r = (int)(i / dc), c = (int)(i % dc);
  dst[i] = (c < sc) ? (_Float16)src[(size_t)r * sc + c] : (_Float16)0.f;
}

__global__ void pack_xin(const float* __restrict__ feat, const float* __restrict__ aux1,
                         const float* __restrict__ x, _Float16* __restrict__ xin) {
  size_t i = (size_t)blockIdx.x * blockDim.x + threadIdx.x;
  if (i >= (size_t)M_ * DINP_) return;
  int m = (int)(i / DINP_), c = (int)(i % DINP_);
  float v = 0.f;
  if (c < 80)       v = feat[(size_t)m * 80 + c];
  else if (c < 144) v = aux1[(size_t)m * 64 + (c - 80)];
  else if (c < 147) v = x[(size_t)m * 3 + (c - 144)];
  xin[i] = (_Float16)v;
}

__global__ void pack_aux(const float* __restrict__ aux2, const float* __restrict__ aux3,
                         _Float16* __restrict__ Ac, _Float16* __restrict__ Af) {
  size_t i = (size_t)blockIdx.x * blockDim.x + threadIdx.x;
  if (i >= (size_t)M_ * AUX_) return;
  int m = (int)(i / AUX_), c = (int)(i % AUX_);
  Ac[(size_t)m * KC_ + HALF_ + c] = (_Float16)aux2[i];
  Af[(size_t)m * KC_ + HALF_ + c] = (_Float16)aux3[i];
}

__global__ void init_k(_Float16* hbuf, unsigned* ctr) {
  int i = blockIdx.x * blockDim.x + threadIdx.x;
  if (i == 0) *ctr = 0u;
  if (i < 2 * B_ * RNN_) hbuf[i] = (_Float16)0.f;
}

// In-place log_softmax over 256 classes: one wave32 per row, 8 regs/lane, shfl_xor reductions.
__global__ __launch_bounds__(256) void logsoftmax256(float* __restrict__ x, int rows) {
  int row = blockIdx.x * (blockDim.x >> 5) + (threadIdx.x >> 5);
  int lane = threadIdx.x & 31;
  if (row >= rows) return;
  float* p = x + (size_t)row * NC_;
  float v[8], mx = -INFINITY;
#pragma unroll
  for (int i = 0; i < 8; ++i) { v[i] = p[lane + i * 32]; mx = fmaxf(mx, v[i]); }
#pragma unroll
  for (int s = 16; s > 0; s >>= 1) mx = fmaxf(mx, __shfl_xor(mx, s, 32));
  float sum = 0.f;
#pragma unroll
  for (int i = 0; i < 8; ++i) sum += expf(v[i] - mx);
#pragma unroll
  for (int s = 16; s > 0; s >>= 1) sum += __shfl_xor(sum, s, 32);
  float lse = mx + logf(sum);
#pragma unroll
  for (int i = 0; i < 8; ++i) p[lane + i * 32] = v[i] - lse;
}

// ---------------- host orchestration ----------------
extern "C" void kernel_launch(void* const* d_in, const int* in_sizes, int n_in,
                              void* d_out, int out_size, void* d_ws, size_t ws_size,
                              hipStream_t stream) {
  const float* x     = (const float*)d_in[0];
  const float* feat  = (const float*)d_in[1];
  const float* aux1  = (const float*)d_in[2];
  const float* aux2  = (const float*)d_in[3];
  const float* aux3  = (const float*)d_in[4];
  const float* W_ih  = (const float*)d_in[5];
  const float* b_ih  = (const float*)d_in[6];
  const float* W_hh  = (const float*)d_in[7];
  const float* b_hh  = (const float*)d_in[8];
  const float* fc1_W = (const float*)d_in[9];
  const float* fc1_b = (const float*)d_in[10];
  const float* fc2_W = (const float*)d_in[11];
  const float* fc2_b = (const float*)d_in[12];
  const float* fc3_W = (const float*)d_in[13];
  const float* fc3_b = (const float*)d_in[14];
  const float* fc4_W = (const float*)d_in[15];
  const float* fc4_b = (const float*)d_in[16];
  float* out = (float*)d_out;

  // Workspace carve-up (256B aligned)
  char* ws = (char*)d_ws;
  size_t off = 0;
  auto take = [&](size_t bytes) { char* p = ws + off; off += (bytes + 255) & ~(size_t)255; return p; };
  unsigned*  ctr    = (unsigned*)take(256);
  _Float16*  hbuf   = (_Float16*)take((size_t)2 * B_ * RNN_ * 2);
  _Float16*  Wih16  = (_Float16*)take((size_t)G3_ * DINP_ * 2);
  _Float16*  Whh16  = (_Float16*)take((size_t)G3_ * RNN_ * 2);
  _Float16*  f1W16  = (_Float16*)take((size_t)FC_ * KC_ * 2);
  _Float16*  f2W16  = (_Float16*)take((size_t)NC_ * FC_ * 2);
  _Float16*  f3W16  = (_Float16*)take((size_t)FC_ * KC_ * 2);
  _Float16*  f4W16  = (_Float16*)take((size_t)NC_ * FC_ * 2);
  _Float16*  xin16  = (_Float16*)take((size_t)M_ * DINP_ * 2);
  _Float16*  xp16   = (_Float16*)take((size_t)M_ * G3_ * 2);
  _Float16*  Ac     = (_Float16*)take((size_t)M_ * KC_ * 2);
  _Float16*  Af     = (_Float16*)take((size_t)M_ * KC_ * 2);
  _Float16*  oc16   = (_Float16*)take((size_t)M_ * FC_ * 2);
  _Float16*  of16   = (_Float16*)take((size_t)M_ * FC_ * 2);

  float* pc_out = out;                               // [M][256]
  float* pf_out = out + (size_t)M_ * NC_;            // [M][256]
  float* hn_out = out + (size_t)2 * M_ * NC_;        // [16][896]

  auto g1 = [](size_t n) { return dim3((unsigned)((n + 255) / 256)); };

  // init + weight conversion + input packing
  init_k<<<g1(2 * B_ * RNN_), 256, 0, stream>>>(hbuf, ctr);
  cvt_pad<<<g1((size_t)G3_ * DINP_), 256, 0, stream>>>(W_ih, Wih16, G3_, 147, DINP_);
  cvt_pad<<<g1((size_t)G3_ * RNN_),  256, 0, stream>>>(W_hh, Whh16, G3_, RNN_, RNN_);
  cvt_pad<<<g1((size_t)FC_ * KC_),   256, 0, stream>>>(fc1_W, f1W16, FC_, KC_, KC_);
  cvt_pad<<<g1((size_t)NC_ * FC_),   256, 0, stream>>>(fc2_W, f2W16, NC_, FC_, FC_);
  cvt_pad<<<g1((size_t)FC_ * KC_),   256, 0, stream>>>(fc3_W, f3W16, FC_, KC_, KC_);
  cvt_pad<<<g1((size_t)NC_ * FC_),   256, 0, stream>>>(fc4_W, f4W16, NC_, FC_, FC_);
  pack_xin<<<g1((size_t)M_ * DINP_), 256, 0, stream>>>(feat, aux1, x, xin16);
  pack_aux<<<g1((size_t)M_ * AUX_),  256, 0, stream>>>(aux2, aux3, Ac, Af);

  const int Mtiles = M_ / 16;  // 4096
  // x_proj = xin @ W_ih^T + b_ih   (f16 out, 65536x2688, K=160)
  {
    int ngr = G3_ / 64, tiles = Mtiles * ngr;
    gemm16<<<dim3((tiles + 7) / 8), 256, 0, stream>>>(xin16, Wih16, b_ih, xp16, nullptr,
                                                      Mtiles, ngr, G3_, DINP_, 0);
  }
  // persistent GRU (LDS-resident W_hh slices, ~176KB dynamic LDS per WGP)
  (void)hipFuncSetAttribute((const void*)gru_kernel,
                            hipFuncAttributeMaxDynamicSharedMemorySize, (int)GRU_SMEM);
  gru_kernel<<<dim3(NWG_), 192, GRU_SMEM, stream>>>(Whh16, b_hh, xp16, hbuf, Ac, Af, hn_out, ctr);

  // coarse head: fc1 (relu) -> fc2 (logits into d_out) -> log_softmax in place
  {
    int ngr = FC_ / 64, tiles = Mtiles * ngr;
    gemm16<<<dim3((tiles + 7) / 8), 256, 0, stream>>>(Ac, f1W16, fc1_b, oc16, nullptr,
                                                      Mtiles, ngr, FC_, KC_, 1);
    int ngr2 = NC_ / 64, tiles2 = Mtiles * ngr2;
    gemm16<<<dim3((tiles2 + 7) / 8), 256, 0, stream>>>(oc16, f2W16, fc2_b, nullptr, pc_out,
                                                       Mtiles, ngr2, NC_, FC_, 0);
    logsoftmax256<<<dim3(M_ / 8), 256, 0, stream>>>(pc_out, M_);
  }
  // fine head: fc3 -> fc4 -> log_softmax
  {
    int ngr = FC_ / 64, tiles = Mtiles * ngr;
    gemm16<<<dim3((tiles + 7) / 8), 256, 0, stream>>>(Af, f3W16, fc3_b, of16, nullptr,
                                                      Mtiles, ngr, FC_, KC_, 1);
    int ngr2 = NC_ / 64, tiles2 = Mtiles * ngr2;
    gemm16<<<dim3((tiles2 + 7) / 8), 256, 0, stream>>>(of16, f4W16, fc4_b, nullptr, pf_out,
                                                       Mtiles, ngr2, NC_, FC_, 0);
    logsoftmax256<<<dim3(M_ / 8), 256, 0, stream>>>(pf_out, M_);
  }
  (void)in_sizes; (void)n_in; (void)out_size; (void)ws_size;
}
